// JointPredReprModule_56934086475869
// MI455X (gfx1250) — matrix operand order repaired and checked by reading, not compiled, not run on hardware
//
#include <hip/hip_runtime.h>
#include <hip/hip_bf16.h>
#include <math.h>

typedef __attribute__((ext_vector_type(16))) _Float16 v16h;
typedef __attribute__((ext_vector_type(8)))  float    v8f;

#define WMMA_F16(a, b, c) \
  __builtin_amdgcn_wmma_f32_16x16x32_f16(false, (a), false, (b), (short)0, (c), false, false)

union Frag16 { v16h v; float4 f4[2]; };

__device__ inline v16h ld_frag_split(const _Float16* p0, const _Float16* p1) {
  Frag16 u;
  u.f4[0] = *(const float4*)p0;
  u.f4[1] = *(const float4*)p1;
  return u.v;
}

// ---------------------------------------------------------------------------
// WMMA GEMM:  C[M,N] = A[M,K] (f16, row-major) * BT[N,K]^T (f16)
// Epilogue behavior fixed at compile time (branch-free codegen).
// M % 128 == 0, N % 128 == 0, K % 64 == 0.
// Block: 256 threads (8 waves), block tile 128x128, wave tile 32x64 (2x4 WMMA),
// LDS double-buffered: one barrier per K-step, global loads overlapped w/ WMMA.
// ---------------------------------------------------------------------------
template <bool HAS_BIAS, bool HAS_RES, bool DO_RELU, bool OUT_F, bool OUT_H>
__global__ __launch_bounds__(256) void gemm_wmma_kernel(
    const _Float16* __restrict__ A, const _Float16* __restrict__ BT,
    const float* __restrict__ bias, const float* __restrict__ res,
    float* __restrict__ outF, _Float16* __restrict__ outH,
    int M, int N, int K)
{
  constexpr int ST = 40;  // LDS row stride (halves): 80B, 16B-aligned, padded
  __shared__ _Float16 As[2][128 * ST];
  __shared__ _Float16 Bs[2][128 * ST];

  const int tid  = threadIdx.x;
  const int wave = tid >> 5, lane = tid & 31;
  const int hl   = lane & 15, hi = lane >> 4;
  const int wr   = wave >> 1, wc = wave & 1;   // 4x2 wave grid
  const int n0   = blockIdx.x * 128;
  const int m0   = blockIdx.y * 128;

  v8f acc[2][4] = {};

  // Staging: 128 rows x 32 halves per tile; 16 halves (2x float4) per thread.
  const int trow = tid >> 1, tcol = (tid & 1) * 16;
  const _Float16* Ag = A  + (size_t)(m0 + trow) * K + tcol;
  const _Float16* Bg = BT + (size_t)(n0 + trow) * K + tcol;
  const int ldsOff = trow * ST + tcol;

  { // preload tile 0
    const float4 a0 = *(const float4*)(Ag);
    const float4 a1 = *(const float4*)(Ag + 8);
    const float4 c0 = *(const float4*)(Bg);
    const float4 c1 = *(const float4*)(Bg + 8);
    *(float4*)&As[0][ldsOff]     = a0;
    *(float4*)&As[0][ldsOff + 8] = a1;
    *(float4*)&Bs[0][ldsOff]     = c0;
    *(float4*)&Bs[0][ldsOff + 8] = c1;
  }
  __syncthreads();

  int buf = 0;
  for (int k0 = 0; k0 < K; k0 += 32, buf ^= 1) {
    const bool hasNext = (k0 + 32) < K;
    float4 a0, a1, c0, c1;
    if (hasNext) {  // issue next tile's global loads before this tile's math
      a0 = *(const float4*)(Ag + k0 + 32);
      a1 = *(const float4*)(Ag + k0 + 40);
      c0 = *(const float4*)(Bg + k0 + 32);
      c1 = *(const float4*)(Bg + k0 + 40);
      if (k0 + 64 < K) {
        __builtin_prefetch(Ag + k0 + 64, 0, 1);   // global_prefetch_b8
        __builtin_prefetch(Bg + k0 + 64, 0, 1);
      }
    }

    v16h af[2], bf[4];
#pragma unroll
    for (int r = 0; r < 2; ++r) {
      // A lane layout: row = hl; K = {hi*8..+7} and {16+hi*8..+7}
      const _Float16* p = &As[buf][(wr * 32 + r * 16 + hl) * ST + hi * 8];
      af[r] = ld_frag_split(p, p + 16);
    }
#pragma unroll
    for (int c = 0; c < 4; ++c) {
      // B lane layout: col = hl; K = {hi*8..+7} and {16+hi*8..+7}
      const _Float16* p = &Bs[buf][(wc * 64 + c * 16 + hl) * ST + hi * 8];
      bf[c] = ld_frag_split(p, p + 16);
    }
#pragma unroll
    for (int r = 0; r < 2; ++r)
#pragma unroll
      for (int c = 0; c < 4; ++c)
        acc[r][c] = WMMA_F16(af[r], bf[c], acc[r][c]);

    if (hasNext) {
      const int nb = buf ^ 1;
      *(float4*)&As[nb][ldsOff]     = a0;
      *(float4*)&As[nb][ldsOff + 8] = a1;
      *(float4*)&Bs[nb][ldsOff]     = c0;
      *(float4*)&Bs[nb][ldsOff + 8] = c1;
    }
    __syncthreads();
  }

#pragma unroll
  for (int r = 0; r < 2; ++r)
#pragma unroll
    for (int c = 0; c < 4; ++c) {
      const int colBase = n0 + wc * 64 + c * 16 + hl;
      const int rowBase = m0 + wr * 32 + r * 16 + hi * 8;
      float bv = 0.f;
      if (HAS_BIAS) bv = bias[colBase];
#pragma unroll
      for (int j = 0; j < 8; ++j) {
        const int row = rowBase + j;
        float v = acc[r][c][j] + bv;
        const size_t o = (size_t)row * N + colBase;
        if (HAS_RES) v += res[o];
        if (DO_RELU) v = fmaxf(v, 0.f);
        if (OUT_F) outF[o] = v;
        if (OUT_H) outH[o] = (_Float16)v;
      }
    }
}

// ---------------------------------------------------------------------------
// Fused attention, one block per (batch, head). F=128, dh=64.
// S = Q K^T * 1/8, block-causal mask, softmax, O = P V.  All WMMA in LDS.
// ---------------------------------------------------------------------------
__global__ __launch_bounds__(256) void attention_kernel(
    const _Float16* __restrict__ Q, const _Float16* __restrict__ Km,
    const _Float16* __restrict__ V, _Float16* __restrict__ O)
{
  constexpr int D = 512, F = 128, H = 8, DH = 64;
  constexpr int QST = 72, VTS = 136, SST = 132, PST = 136;
  extern __shared__ unsigned char smemRaw[];
  _Float16* Qs = (_Float16*)smemRaw;          // [128][72]
  _Float16* Ks = Qs + 128 * QST;              // [128][72]
  _Float16* Vt = Ks + 128 * QST;              // [64][136]  (V transposed: [dh][F])
  float*    Sf = (float*)(Vt + 64 * VTS);     // [128][132]
  _Float16* Ps = (_Float16*)(Sf + 128 * SST); // [128][136]

  const int b = blockIdx.x / H, h = blockIdx.x % H;
  const int tid = threadIdx.x;
  const size_t base = ((size_t)b * F) * D + (size_t)h * DH;

  for (int idx = tid; idx < 1024; idx += 256) {
    const int f = idx >> 3, dg = (idx & 7) * 8;
    *(float4*)&Qs[f * QST + dg] = *(const float4*)(Q  + base + (size_t)f * D + dg);
    *(float4*)&Ks[f * QST + dg] = *(const float4*)(Km + base + (size_t)f * D + dg);
    union { float4 f4; _Float16 h[8]; } u;
    u.f4 = *(const float4*)(V + base + (size_t)f * D + dg);
#pragma unroll
    for (int i = 0; i < 8; ++i) Vt[(dg + i) * VTS + f] = u.h[i];
  }
  __syncthreads();

  const int wave = tid >> 5, lane = tid & 31;
  const int hl = lane & 15, hi = lane >> 4;
  const int rBase = wave * 16;  // each wave owns a 16-row strip

  { // ----- S = Q K^T (M=128, N=128, K=64) -----
    v8f acc[8] = {};
#pragma unroll
    for (int kk = 0; kk < 64; kk += 32) {
      const _Float16* qp = &Qs[(rBase + hl) * QST + kk + hi * 8];
      const v16h af = ld_frag_split(qp, qp + 16);
#pragma unroll
      for (int ct = 0; ct < 8; ++ct) {
        const _Float16* kp = &Ks[(ct * 16 + hl) * QST + kk + hi * 8];
        const v16h bf = ld_frag_split(kp, kp + 16);
        acc[ct] = WMMA_F16(af, bf, acc[ct]);
      }
    }
    const float scale = 0.125f;  // 1/sqrt(64)
#pragma unroll
    for (int ct = 0; ct < 8; ++ct) {
      const int col = ct * 16 + hl;
      const int tc = (col & 31) >> 1;
#pragma unroll
      for (int j = 0; j < 8; ++j) {
        const int row = rBase + hi * 8 + j;
        const int tr = (row & 31) >> 1;
        const float v = acc[ct][j] * scale;
        Sf[row * SST + col] = (tr <= tc) ? v : -__builtin_inff();
      }
    }
  }
  __syncthreads();

  if (tid < 128) { // ----- softmax over row tid -----
    float* rowp = &Sf[tid * SST];
    float mx = -__builtin_inff();
    for (int j = 0; j < 128; ++j) mx = fmaxf(mx, rowp[j]);
    float s = 0.f;
    for (int j = 0; j < 128; ++j) { const float e = __expf(rowp[j] - mx); rowp[j] = e; s += e; }
    const float inv = 1.f / s;
    _Float16* pp = &Ps[tid * PST];
    for (int j = 0; j < 128; ++j) pp[j] = (_Float16)(rowp[j] * inv);
  }
  __syncthreads();

  { // ----- O = P V (M=128, N=64, K=128); BT = V^T already staged -----
    v8f acc[4] = {};
#pragma unroll
    for (int kk = 0; kk < 128; kk += 32) {
      const _Float16* pp = &Ps[(rBase + hl) * PST + kk + hi * 8];
      const v16h af = ld_frag_split(pp, pp + 16);
#pragma unroll
      for (int ct = 0; ct < 4; ++ct) {
        const _Float16* vp = &Vt[(ct * 16 + hl) * VTS + kk + hi * 8];
        const v16h bf = ld_frag_split(vp, vp + 16);
        acc[ct] = WMMA_F16(af, bf, acc[ct]);
      }
    }
#pragma unroll
    for (int ct = 0; ct < 4; ++ct) {
      const int col = ct * 16 + hl;
#pragma unroll
      for (int j = 0; j < 8; ++j) {
        const int row = rBase + hi * 8 + j;
        O[base + (size_t)row * D + col] = (_Float16)acc[ct][j];
      }
    }
  }
}

// ---------------------------------------------------------------------------
// Embedding assembly: x[b, a*32+2l(+1), :] = (obs | onehot@actW+actb) + pos + seg
// One block per (l,b,a); 256 threads x 2 columns.
// ---------------------------------------------------------------------------
__global__ __launch_bounds__(256) void embed_kernel(
    const float* __restrict__ obs, const float* __restrict__ oh,
    const float* __restrict__ actW, const float* __restrict__ actB,
    const float* __restrict__ seg, const float* __restrict__ pos,
    float* __restrict__ x, _Float16* __restrict__ x16)
{
  constexpr int Bn = 128, An = 4, Ln = 16, Dn = 512, Fn = 128, ACTn = 16;
  __shared__ float ohs[ACTn];
  const int blk = blockIdx.x;             // linear over (l, b, a)
  const int l = blk / (Bn * An);
  const int rem = blk % (Bn * An);
  const int b = rem / An, a = rem % An;
  const int tid = threadIdx.x;
  if (tid < ACTn) ohs[tid] = oh[(size_t)blk * ACTn + tid];
  __syncthreads();

  const int fo = a * 2 * Ln + 2 * l;
  const size_t obsBase = (size_t)blk * Dn;
  const size_t xo = ((size_t)b * Fn + fo) * Dn;
  const size_t xa = xo + Dn;
#pragma unroll
  for (int s = 0; s < 2; ++s) {
    const int d = tid + s * 256;
    const float pe = pos[l * Dn + d] + seg[a * Dn + d];
    const float ov = obs[obsBase + d] + pe;
    float av = actB[d];
#pragma unroll
    for (int c = 0; c < ACTn; ++c) av += ohs[c] * actW[c * Dn + d];
    av += pe;
    x[xo + d] = ov; x16[xo + d] = (_Float16)ov;
    x[xa + d] = av; x16[xa + d] = (_Float16)av;
  }
}

// ---------------------------------------------------------------------------
// Weight conversion f32 -> f16 transposed: Wt[n,k] = W[k,n]
// ---------------------------------------------------------------------------
__global__ void convT_kernel(const float* __restrict__ W, _Float16* __restrict__ Wt,
                             int K, int N)
{
  const int idx = blockIdx.x * 256 + threadIdx.x;
  const int n = idx / K, k = idx % K;
  Wt[(size_t)n * K + k] = (_Float16)W[(size_t)k * N + n];
}

// ---------------------------------------------------------------------------
// BatchNorm helpers (training-mode batch stats over R=16384 rows, D=512 cols)
// ---------------------------------------------------------------------------
__global__ void bn_zero_kernel(float* p) { p[blockIdx.x * 256 + threadIdx.x] = 0.f; }

__global__ __launch_bounds__(256) void bn_stats_kernel(
    const float* __restrict__ x, float* __restrict__ sum, float* __restrict__ sumsq)
{
  const int c0 = threadIdx.x;            // columns c0 and c0+256
  const int r0 = blockIdx.x * 64;        // 64 rows per block
  float s0 = 0.f, q0 = 0.f, s1 = 0.f, q1 = 0.f;
  for (int r = 0; r < 64; ++r) {
    const float* row = x + (size_t)(r0 + r) * 512;
    const float v = row[c0];       s0 += v; q0 += v * v;
    const float w = row[c0 + 256]; s1 += w; q1 += w * w;
  }
  atomicAdd(&sum[c0], s0);        atomicAdd(&sumsq[c0], q0);
  atomicAdd(&sum[c0 + 256], s1);  atomicAdd(&sumsq[c0 + 256], q1);
}

__global__ void bn_finalize_kernel(const float* __restrict__ sum, const float* __restrict__ sumsq,
                                   const float* __restrict__ g, const float* __restrict__ beta,
                                   float* __restrict__ scale, float* __restrict__ shift,
                                   float invR)
{
  const int c = threadIdx.x;  // 512 threads
  const float m = sum[c] * invR;
  const float var = sumsq[c] * invR - m * m;
  const float sc = g[c] * rsqrtf(var + 1e-5f);
  scale[c] = sc;
  shift[c] = beta[c] - m * sc;
}

__global__ __launch_bounds__(256) void bn_apply_kernel(
    float* __restrict__ x, _Float16* __restrict__ x16,
    const float* __restrict__ scale, const float* __restrict__ shift)
{
  const size_t idx = (size_t)blockIdx.x * 256 + threadIdx.x;
  const int c = (int)(idx & 511);
  const float v = x[idx] * scale[c] + shift[c];
  x[idx] = v;
  x16[idx] = (_Float16)v;
}

// out[b,i,d] = x[b, 2i, d]
__global__ __launch_bounds__(256) void gather_kernel(
    const float* __restrict__ x, float* __restrict__ out)
{
  const size_t idx = (size_t)blockIdx.x * 256 + threadIdx.x;
  const int d = (int)(idx & 511);
  const size_t r = idx >> 9;            // b*64 + i
  const size_t b = r >> 6;
  const int i = (int)(r & 63);
  out[idx] = x[(((size_t)b * 128) + 2 * i) * 512 + d];
}

// ---------------------------------------------------------------------------
extern "C" void kernel_launch(void* const* d_in, const int* in_sizes, int n_in,
                              void* d_out, int out_size, void* d_ws, size_t ws_size,
                              hipStream_t stream)
{
  (void)in_sizes; (void)n_in; (void)out_size; (void)ws_size;
  constexpr int L = 16, B = 128, A = 4, D = 512, H = 8, F = 128;
  constexpr int R = B * F;  // 16384 rows

  const float* obs  = (const float*)d_in[0];
  const float* oh   = (const float*)d_in[1];
  const float* actW = (const float*)d_in[2];
  const float* actB = (const float*)d_in[3];
  const float* seg  = (const float*)d_in[4];
  const float* pos  = (const float*)d_in[5];
  const float* Wq   = (const float*)d_in[6];
  const float* Wk   = (const float*)d_in[7];
  const float* Wv   = (const float*)d_in[8];
  const float* Wc   = (const float*)d_in[9];
  const float* bc   = (const float*)d_in[10];
  const float* W1   = (const float*)d_in[11];
  const float* b1   = (const float*)d_in[12];
  const float* W2   = (const float*)d_in[13];
  const float* b2   = (const float*)d_in[14];
  const float* g1   = (const float*)d_in[15];
  const float* be1  = (const float*)d_in[16];
  const float* g2   = (const float*)d_in[17];
  const float* be2  = (const float*)d_in[18];
  // d_in[19] = mask: recomputed analytically in-kernel.

  char* ws = (char*)d_ws;
  size_t off = 0;
  auto alloc = [&](size_t bytes) -> void* {
    void* p = ws + off;
    off = (off + bytes + 255) & ~(size_t)255;
    return p;
  };
  float*    x     = (float*)   alloc((size_t)R * D * 4);
  _Float16* x16   = (_Float16*)alloc((size_t)R * D * 2);
  _Float16* q16   = (_Float16*)alloc((size_t)R * D * 2);
  _Float16* k16   = (_Float16*)alloc((size_t)R * D * 2);
  _Float16* v16   = (_Float16*)alloc((size_t)R * D * 2);
  _Float16* att16 = (_Float16*)alloc((size_t)R * D * 2);
  _Float16* mid16 = (_Float16*)alloc((size_t)R * 4 * D * 2);
  _Float16* wqT   = (_Float16*)alloc((size_t)4 * D * D * 2);
  _Float16* wkT   = (_Float16*)alloc((size_t)4 * D * D * 2);
  _Float16* wvT   = (_Float16*)alloc((size_t)4 * D * D * 2);
  _Float16* wcT   = (_Float16*)alloc((size_t)4 * D * D * 2);
  _Float16* w1T   = (_Float16*)alloc((size_t)4 * D * 4 * D * 2);
  _Float16* w2T   = (_Float16*)alloc((size_t)4 * D * 4 * D * 2);
  float* bnSum    = (float*)alloc(512 * 4);
  float* bnSq     = (float*)alloc(512 * 4);
  float* bnScale  = (float*)alloc(512 * 4);
  float* bnShift  = (float*)alloc(512 * 4);

  // ---- convert + transpose all weights to f16 (N x K) ----
  for (int i = 0; i < 4; ++i) {
    const size_t wO = (size_t)i * D * D;
    const size_t fO = (size_t)i * D * 4 * D;
    convT_kernel<<<D * D / 256, 256, 0, stream>>>(Wq + wO, wqT + wO, D, D);
    convT_kernel<<<D * D / 256, 256, 0, stream>>>(Wk + wO, wkT + wO, D, D);
    convT_kernel<<<D * D / 256, 256, 0, stream>>>(Wv + wO, wvT + wO, D, D);
    convT_kernel<<<D * D / 256, 256, 0, stream>>>(Wc + wO, wcT + wO, D, D);
    convT_kernel<<<D * 4 * D / 256, 256, 0, stream>>>(W1 + fO, w1T + fO, D, 4 * D);
    convT_kernel<<<D * 4 * D / 256, 256, 0, stream>>>(W2 + fO, w2T + fO, 4 * D, D);
  }

  // ---- embedding assembly ----
  embed_kernel<<<L * B * A, 256, 0, stream>>>(obs, oh, actW, actB, seg, pos, x, x16);

  constexpr int attSmem =
      (128 * 72 + 128 * 72 + 64 * 136) * 2 + 128 * 132 * 4 + 128 * 136 * 2;  // 156,672 B
  (void)hipFuncSetAttribute(reinterpret_cast<const void*>(attention_kernel),
                            hipFuncAttributeMaxDynamicSharedMemorySize, attSmem);

  const dim3 gQKV(D / 128, R / 128);       // (4, 128)
  const dim3 gFF1(4 * D / 128, R / 128);   // (16, 128)

  auto batchnorm = [&](const float* g, const float* beta) {
    bn_zero_kernel<<<2, 256, 0, stream>>>(bnSum);
    bn_zero_kernel<<<2, 256, 0, stream>>>(bnSq);
    bn_stats_kernel<<<R / 64, 256, 0, stream>>>(x, bnSum, bnSq);
    bn_finalize_kernel<<<1, 512, 0, stream>>>(bnSum, bnSq, g, beta, bnScale, bnShift, 1.f / R);
    bn_apply_kernel<<<(int)((size_t)R * D / 256), 256, 0, stream>>>(x, x16, bnScale, bnShift);
  };

  for (int i = 0; i < 4; ++i) {
    const size_t wO = (size_t)i * D * D;
    const size_t fO = (size_t)i * D * 4 * D;

    // Q, K, V projections (f16 out only)
    gemm_wmma_kernel<false, false, false, false, true><<<gQKV, 256, 0, stream>>>(
        x16, wqT + wO, nullptr, nullptr, nullptr, q16, R, D, D);
    gemm_wmma_kernel<false, false, false, false, true><<<gQKV, 256, 0, stream>>>(
        x16, wkT + wO, nullptr, nullptr, nullptr, k16, R, D, D);
    gemm_wmma_kernel<false, false, false, false, true><<<gQKV, 256, 0, stream>>>(
        x16, wvT + wO, nullptr, nullptr, nullptr, v16, R, D, D);
    // fused attention per (b, h)
    attention_kernel<<<B * H, 256, attSmem, stream>>>(q16, k16, v16, att16);
    // output projection: bias + residual into x (f32)
    gemm_wmma_kernel<true, true, false, true, false><<<gQKV, 256, 0, stream>>>(
        att16, wcT + wO, bc + (size_t)i * D, x, x, nullptr, R, D, D);
    batchnorm(g1 + (size_t)i * D, be1 + (size_t)i * D);
    // FFN: relu(x@W1+b1) -> f16 mid; mid@W2+b2 + residual -> f32 x
    gemm_wmma_kernel<true, false, true, false, true><<<gFF1, 256, 0, stream>>>(
        x16, w1T + fO, b1 + (size_t)i * 4 * D, nullptr, nullptr, mid16, R, 4 * D, D);
    gemm_wmma_kernel<true, true, false, true, false><<<gQKV, 256, 0, stream>>>(
        mid16, w2T + fO, b2 + (size_t)i * D, x, x, nullptr, R, D, 4 * D);
    batchnorm(g2 + (size_t)i * D, be2 + (size_t)i * D);
  }

  // gather obs positions
  gather_kernel<<<(int)((size_t)B * 64 * D / 256), 256, 0, stream>>>(x, (float*)d_out);
}
